// UniSeqAttentionBlock_81114752353009
// MI455X (gfx1250) — matrix operand order repaired and checked by reading, not compiled, not run on hardware
//
#include <hip/hip_runtime.h>
#include <hip/hip_bf16.h>

// ---------------------------------------------------------------------------
// UniSeqAttentionBlock for MI455X (gfx1250, wave32, WMMA bf16 / f32-acc)
// b=4, t=512, m=8, d=64, p=128, heads h=4, e=32
// ---------------------------------------------------------------------------

typedef __attribute__((ext_vector_type(16))) __bf16 v16bf;
typedef __attribute__((ext_vector_type(8)))  float  v8f;

#define B_  4
#define T_  512
#define M_  8
#define D_  64
#define P_  128
#define H_  4
#define E_  32

// workspace element counts (bf16)
#define QK_ELEMS  ((size_t)B_ * M_ * H_ * T_ * E_)   // 2,097,152
#define TT_ELEMS  ((size_t)B_ * H_ * T_ * E_)        // 262,144

// ---------------- Kernel 1: fused projection GEMM (X @ W + b) --------------
// mode 0: Q/K layout [b][m][h][t][e]   rowsPerB = t*m = 4096
// mode 1: V  layout [b][m][h][e][t]   (transposed for PV B-fragment)
// mode 2: Qt/Kt layout [b][h][t][e]   rowsPerB = t = 512
#define AS_STRIDE 72
#define WT_STRIDE 72

__global__ __launch_bounds__(256)
void uniseq_proj_kernel(const float* __restrict__ X, const float* __restrict__ W,
                        const float* __restrict__ bias, __bf16* __restrict__ dst,
                        int rowsPerB, int mode) {
    __shared__ __bf16 As[16 * AS_STRIDE];
    __shared__ __bf16 Wt[P_ * WT_STRIDE];

    const int strip = blockIdx.x;
    const int bi    = blockIdx.y;
    const int tid   = threadIdx.x;
    const int lane  = tid & 31;
    const int wave  = tid >> 5;

    // --- stage activations (16x64 f32 -> bf16) and W (64x128 -> 128x64^T) ---
    {
        const float* xrow = X + ((size_t)bi * rowsPerB + strip * 16) * D_;
#pragma unroll
        for (int i = 0; i < 4; ++i) {
            int idx = tid + i * 256;                   // 0..1023
            As[(idx >> 6) * AS_STRIDE + (idx & 63)] = (__bf16)xrow[idx];
        }
#pragma unroll
        for (int i = 0; i < 32; ++i) {
            int idx = tid + i * 256;                   // 0..8191, k=idx/128, n=idx%128
            int k2 = idx >> 7, n2 = idx & 127;
            Wt[n2 * WT_STRIDE + k2] = (__bf16)W[idx];
        }
    }
    __syncthreads();

    const int lrow  = lane & 15;
    const int lhalf = lane >> 4;
    const int akb   = lhalf * 8;    // A-frag K base within 16
    const int bkb   = lhalf * 16;   // B-frag K base within 32
    const int n0    = wave * 16;    // this wave's output column tile

    // A fragments (16x32 bf16, two k-steps of 32)
    v16bf a0, a1, b0, b1;
    const __bf16* arow = As + lrow * AS_STRIDE;
    ((uint4*)&a0)[0] = *(const uint4*)(arow + akb);
    ((uint4*)&a0)[1] = *(const uint4*)(arow + 16 + akb);
    ((uint4*)&a1)[0] = *(const uint4*)(arow + 32 + akb);
    ((uint4*)&a1)[1] = *(const uint4*)(arow + 48 + akb);
    const __bf16* wcol = Wt + (n0 + lrow) * WT_STRIDE;
    ((uint4*)&b0)[0] = *(const uint4*)(wcol + akb);
    ((uint4*)&b0)[1] = *(const uint4*)(wcol + 16 + akb);
    ((uint4*)&b1)[0] = *(const uint4*)(wcol + 32 + akb);
    ((uint4*)&b1)[1] = *(const uint4*)(wcol + 48 + akb);

    v8f c = {};
    c = __builtin_amdgcn_wmma_f32_16x16x32_bf16(false, a0, false, b0, (short)0, c, false, false);
    c = __builtin_amdgcn_wmma_f32_16x16x32_bf16(false, a1, false, b1, (short)0, c, false, false);

    const int   cn = n0 + lrow;
    const float bv = bias[cn];
    const int   hh = cn >> 5, ee = cn & 31;
#pragma unroll
    for (int j = 0; j < 8; ++j) {
        const int   R   = strip * 16 + j + lhalf * 8;
        const float val = c[j] + bv;
        size_t idx;
        if (mode == 0) {
            const int ti = R >> 3, mi = R & 7;         // R = ti*m + mi
            idx = ((((size_t)bi * M_ + mi) * H_ + hh) * T_ + ti) * E_ + ee;
        } else if (mode == 1) {
            const int ti = R >> 3, mi = R & 7;
            idx = ((((size_t)bi * M_ + mi) * H_ + hh) * E_ + ee) * T_ + ti;
        } else {
            idx = (((size_t)bi * H_ + hh) * T_ + R) * E_ + ee;
        }
        dst[idx] = (__bf16)val;
    }
}

// ---------------- Kernel 2: flash attention per (b, m, h) ------------------
#define KS_STRIDE  40    // padded row stride (bf16) for K / Kt panels
#define VT_STRIDE  528   // padded row stride (bf16) for V^T panel
#define PL_STRIDE  40    // P staging row stride

#define LDS_KS_OFF   0
#define LDS_KT_OFF   (T_ * KS_STRIDE)                      // 20480
#define LDS_VT_OFF   (LDS_KT_OFF + T_ * KS_STRIDE)         // 40960
#define LDS_PL_OFF   (LDS_VT_OFF + E_ * VT_STRIDE)         // 57856
#define LDS_MSK_OFF  (LDS_PL_OFF + 8 * 16 * PL_STRIDE)     // 62976 (bf16 elems)
#define ATTN_LDS_BYTES (LDS_MSK_OFF * 2 + T_)              // 126464 bytes

__global__ __launch_bounds__(256)
void uniseq_attn_kernel(const __bf16* __restrict__ Q,  const __bf16* __restrict__ K,
                        const __bf16* __restrict__ Vt, const __bf16* __restrict__ Qt,
                        const __bf16* __restrict__ Kt, const unsigned char* __restrict__ mask,
                        float* __restrict__ out) {
    const int hh = blockIdx.x, mi = blockIdx.y, bi = blockIdx.z;
    const int tid = threadIdx.x, lane = tid & 31, wave = tid >> 5;

    extern __shared__ __bf16 smem[];
    __bf16* Ks_s  = smem + LDS_KS_OFF;
    __bf16* Kts_s = smem + LDS_KT_OFF;
    __bf16* Vts_s = smem + LDS_VT_OFF;
    __bf16* Pls_s = smem + LDS_PL_OFF;
    unsigned char* Msk = (unsigned char*)(smem + LDS_MSK_OFF);

    const int bmh = (bi * M_ + mi) * H_ + hh;
    const int bh  = bi * H_ + hh;

    // ---- cooperative preload of K, Kt (row-major) and V^T panels ----
    for (int r = tid; r < T_; r += 256) {
        const uint4* sk  = (const uint4*)(K  + (size_t)bmh * T_ * E_ + r * E_);
        const uint4* skt = (const uint4*)(Kt + (size_t)bh  * T_ * E_ + r * E_);
        uint4* dk  = (uint4*)(Ks_s  + r * KS_STRIDE);
        uint4* dkt = (uint4*)(Kts_s + r * KS_STRIDE);
#pragma unroll
        for (int j = 0; j < 4; ++j) { dk[j] = sk[j]; dkt[j] = skt[j]; }
    }
    {   // V^T: 32 rows of 512 elems; 8 threads per row, 8 uint4 each
        int i = tid;                                  // exactly 32*8 = 256 chunks
        int r = i >> 3, chunk = i & 7;
        const uint4* sv = (const uint4*)(Vt + (size_t)bmh * T_ * E_ + r * T_ + chunk * 64);
        uint4* dv = (uint4*)(Vts_s + r * VT_STRIDE + chunk * 64);
#pragma unroll
        for (int j = 0; j < 8; ++j) dv[j] = sv[j];
    }
    for (int i = tid; i < T_; i += 256)
        Msk[i] = mask[((size_t)bi * T_ + i) * M_ + mi];
    __syncthreads();

    const int lrow  = lane & 15;
    const int lhalf = lane >> 4;
    const int akb   = lhalf * 8;
    const int bkb   = lhalf * 16;
    const float INF   = __builtin_inff();
    const float SCALE = 0.08838834764831845f;          // 1 / (2*sqrt(32))

    for (int qt_i = 0; qt_i < 4; ++qt_i) {
        const int qtile = wave + qt_i * 8;             // wave-uniform

        // Q / Qt A-fragments: loop-invariant over key tiles
        v16bf qa, qta;
        {
            const __bf16* qrow  = Q  + (size_t)bmh * T_ * E_ + (qtile * 16 + lrow) * E_;
            const __bf16* qtrow = Qt + (size_t)bh  * T_ * E_ + (qtile * 16 + lrow) * E_;
            ((uint4*)&qa)[0]  = *(const uint4*)(qrow + akb);
            ((uint4*)&qa)[1]  = *(const uint4*)(qrow + 16 + akb);
            ((uint4*)&qta)[0] = *(const uint4*)(qtrow + akb);
            ((uint4*)&qta)[1] = *(const uint4*)(qtrow + 16 + akb);
        }

        v8f c0 = {}, c1 = {};
        float mstate[8], lstate[8];
#pragma unroll
        for (int j = 0; j < 8; ++j) { mstate[j] = -INF; lstate[j] = 0.f; }

        const int nkp = (qtile + 2) >> 1;              // causal key-pair count
        for (int kp = 0; kp < nkp; ++kp) {
            const int keybase = kp * 32;
            float sa[2][8];
#pragma unroll
            for (int s16 = 0; s16 < 2; ++s16) {
                const int koff = keybase + s16 * 16;
                const int key  = koff + lrow;
                v16bf kb, ktb;
                const __bf16* ksrc  = Ks_s  + key * KS_STRIDE + akb;
                const __bf16* ktsrc = Kts_s + key * KS_STRIDE + akb;
                ((uint4*)&kb)[0]  = *(const uint4*)(ksrc);
                ((uint4*)&kb)[1]  = *(const uint4*)(ksrc + 16);
                ((uint4*)&ktb)[0] = *(const uint4*)(ktsrc);
                ((uint4*)&ktb)[1] = *(const uint4*)(ktsrc + 16);

                v8f cs = {};
                cs = __builtin_amdgcn_wmma_f32_16x16x32_bf16(false, qta, false, ktb, (short)0, cs, false, false);
                cs = __builtin_amdgcn_wmma_f32_16x16x32_bf16(false, qa,  false, kb,  (short)0, cs, false, false);

                const bool kvalid = (Msk[key] != 0);
#pragma unroll
                for (int j = 0; j < 8; ++j) {
                    const int row = qtile * 16 + j + lhalf * 8;
                    float s = cs[j] * SCALE;
                    if (key > row || !kvalid) s = -INF;
                    sa[s16][j] = s;
                }
            }
            // ---- online softmax over this 32-key group (row = j + 8*lhalf) ----
#pragma unroll
            for (int j = 0; j < 8; ++j) {
                float mloc = fmaxf(sa[0][j], sa[1][j]);
#pragma unroll
                for (int d = 1; d < 16; d <<= 1)
                    mloc = fmaxf(mloc, __shfl_xor(mloc, d, 32));
                const float mold = mstate[j];
                const float mnew = fmaxf(mold, mloc);
                mstate[j] = mnew;
                const float mref  = (mnew == -INF) ? 0.f : mnew;
                const float alpha = (mold == -INF) ? 0.f : __expf(mold - mref);
                const float p0 = __expf(sa[0][j] - mref);
                const float p1 = __expf(sa[1][j] - mref);
                float rsum = p0 + p1;
#pragma unroll
                for (int d = 1; d < 16; d <<= 1)
                    rsum += __shfl_xor(rsum, d, 32);
                lstate[j] = lstate[j] * alpha + rsum;
                c0[j] *= alpha;
                c1[j] *= alpha;
                __bf16* prow = Pls_s + wave * (16 * PL_STRIDE) + (j + lhalf * 8) * PL_STRIDE;
                prow[lrow]      = (__bf16)p0;
                prow[16 + lrow] = (__bf16)p1;
            }
            // in-wave LDS transpose hazard: wait for our own ds stores
            asm volatile("s_wait_dscnt 0x0" ::: "memory");

            // ---- O += P @ V (two 16-col halves of e) ----
            v16bf pa, vb0, vb1;
            const __bf16* psrc = Pls_s + wave * (16 * PL_STRIDE) + lrow * PL_STRIDE;
            ((uint4*)&pa)[0] = *(const uint4*)(psrc + akb);
            ((uint4*)&pa)[1] = *(const uint4*)(psrc + 16 + akb);
            const __bf16* v0src = Vts_s + lrow        * VT_STRIDE + keybase + akb;
            const __bf16* v1src = Vts_s + (16 + lrow) * VT_STRIDE + keybase + akb;
            ((uint4*)&vb0)[0] = *(const uint4*)(v0src);
            ((uint4*)&vb0)[1] = *(const uint4*)(v0src + 16);
            ((uint4*)&vb1)[0] = *(const uint4*)(v1src);
            ((uint4*)&vb1)[1] = *(const uint4*)(v1src + 16);
            c0 = __builtin_amdgcn_wmma_f32_16x16x32_bf16(false, pa, false, vb0, (short)0, c0, false, false);
            c1 = __builtin_amdgcn_wmma_f32_16x16x32_bf16(false, pa, false, vb1, (short)0, c1, false, false);
        }

        // ---- epilogue: normalize and store (b, t, m, p) f32 ----
#pragma unroll
        for (int j = 0; j < 8; ++j) {
            const int   ti  = qtile * 16 + j + lhalf * 8;
            const float l   = lstate[j];
            const float rcp = (l > 0.f) ? __frcp_rn(l) : 0.f;
            float* orow = out + (((size_t)bi * T_ + ti) * M_ + mi) * P_ + hh * E_;
            orow[lrow]      = c0[j] * rcp;
            orow[16 + lrow] = c1[j] * rcp;
        }
    }
}

// ---------------------------------------------------------------------------
extern "C" void kernel_launch(void* const* d_in, const int* in_sizes, int n_in,
                              void* d_out, int out_size, void* d_ws, size_t ws_size,
                              hipStream_t stream) {
    (void)in_sizes; (void)n_in; (void)out_size; (void)ws_size;
    const float* inp  = (const float*)d_in[0];
    const float* pos  = (const float*)d_in[1];
    const unsigned char* mask = (const unsigned char*)d_in[2];
    const float* Wq  = (const float*)d_in[3];
    const float* bq  = (const float*)d_in[4];
    const float* Wk  = (const float*)d_in[5];
    const float* bk  = (const float*)d_in[6];
    const float* Wv  = (const float*)d_in[7];
    const float* bv  = (const float*)d_in[8];
    const float* Wqt = (const float*)d_in[9];
    const float* bqt = (const float*)d_in[10];
    const float* Wkt = (const float*)d_in[11];
    const float* bkt = (const float*)d_in[12];
    float* out = (float*)d_out;

    __bf16* qws  = (__bf16*)d_ws;               // [b][m][h][t][e]
    __bf16* kws  = qws  + QK_ELEMS;             // [b][m][h][t][e]
    __bf16* vws  = kws  + QK_ELEMS;             // [b][m][h][e][t]  (transposed)
    __bf16* qtws = vws  + QK_ELEMS;             // [b][h][t][e]
    __bf16* ktws = qtws + TT_ELEMS;             // [b][h][t][e]

    dim3 blk(256);
    uniseq_proj_kernel<<<dim3(256, B_), blk, 0, stream>>>(inp, Wq,  bq,  qws,  T_ * M_, 0);
    uniseq_proj_kernel<<<dim3(256, B_), blk, 0, stream>>>(inp, Wk,  bk,  kws,  T_ * M_, 0);
    uniseq_proj_kernel<<<dim3(256, B_), blk, 0, stream>>>(inp, Wv,  bv,  vws,  T_ * M_, 1);
    uniseq_proj_kernel<<<dim3(32,  B_), blk, 0, stream>>>(pos, Wqt, bqt, qtws, T_,      2);
    uniseq_proj_kernel<<<dim3(32,  B_), blk, 0, stream>>>(pos, Wkt, bkt, ktws, T_,      2);

    uniseq_attn_kernel<<<dim3(H_, M_, B_), blk, ATTN_LDS_BYTES, stream>>>(
        qws, kws, vws, qtws, ktws, mask, out);
}
